// GCN_47433618817355
// MI455X (gfx1250) — compile-verified, matrix operand order checked by reading, not run on hardware
//
#include <hip/hip_runtime.h>
#include <math.h>

// Problem constants (match reference)
#define N_NODES   100000
#define N_EDGES   1600000
#define NFEAT     512
#define HIDDEN    128
#define NCLASS    40
#define NCLASS_PAD 48   // pad 40 -> 3 WMMA column tiles of 16

typedef __attribute__((ext_vector_type(16))) _Float16 v16h;
typedef __attribute__((ext_vector_type(8)))  float    v8f;

// ---------------------------------------------------------------------------
// Pre-pack W1 (512x128 f32, row-major) into f16 WMMA B-fragment order:
//   Bpack[kt][n][h][i], kt = k-tile (32 K each), n = column, h = lane>=16,
//   i = half index; element = W1[kt*32 + h*8 + (i<8 ? i : i+8)][n].
// A lane then loads its entire v16h fragment as one contiguous 32B read.
// ---------------------------------------------------------------------------
__global__ void gcn_pack_w1(const float* __restrict__ W1,
                            _Float16* __restrict__ out) {
  int idx = blockIdx.x * blockDim.x + threadIdx.x;     // < 16*128*2*16 = 65536
  int i  = idx & 15;
  int h  = (idx >> 4) & 1;
  int n  = (idx >> 5) & (HIDDEN - 1);
  int kt = idx >> 12;
  int k  = kt * 32 + h * 8 + (i < 8 ? i : i + 8);
  out[idx] = (_Float16)W1[(size_t)k * HIDDEN + n];
}

// Same packing for W2 (128x40), zero-padded to 48 columns: [4][48][2][16]
__global__ void gcn_pack_w2(const float* __restrict__ W2,
                            _Float16* __restrict__ out) {
  int idx = blockIdx.x * blockDim.x + threadIdx.x;     // < 4*48*2*16 = 6144
  int i  = idx & 15;
  int h  = (idx >> 4) & 1;
  int c  = idx >> 5;            // kt*48 + n
  int n  = c % NCLASS_PAD;
  int kt = c / NCLASS_PAD;
  int k  = kt * 32 + h * 8 + (i < 8 ? i : i + 8);
  out[idx] = (n < NCLASS) ? (_Float16)W2[(size_t)k * NCLASS + n] : (_Float16)0.f;
}

// ---------------------------------------------------------------------------
// GEMM1: s1[N,128] = x[N,512] @ W1[512,128]   (f16 WMMA, f32 accumulate)
// One wave = one 16x16 tile; block = 8 waves = all 8 HIDDEN column tiles.
// Grid = N/16 = 6250 row tiles (exact). B fragments come pre-packed (two
// global_load_b128 per k-step); A converts fp32->f16 in-register.
// ---------------------------------------------------------------------------
__global__ __launch_bounds__(256) void gcn_gemm1_wmma(
    const float* __restrict__ x, const _Float16* __restrict__ w1p,
    float* __restrict__ s1) {
  const int lane   = threadIdx.x & 31;
  const int wave   = threadIdx.x >> 5;          // 0..7 -> column tile
  const int tile_m = blockIdx.x;                // 0..6249
  const int mn     = lane & 15;
  const int half   = lane >> 4;                 // K sub-stripe select
  const int khalf  = half * 8;
  const int m      = tile_m * 16 + mn;
  const int n      = wave * 16 + mn;
  const v16h* bp   = (const v16h*)w1p;

  v8f c = {};
  for (int kt = 0; kt < NFEAT / 32; ++kt) {
    v16h a;
    const float* xr = x + (size_t)m * NFEAT + kt * 32 + khalf;
#pragma unroll
    for (int i = 0; i < 8; ++i) {
      a[i]     = (_Float16)xr[i];
      a[i + 8] = (_Float16)xr[i + 16];
    }
    v16h b = bp[((size_t)kt * HIDDEN + n) * 2 + half];
    c = __builtin_amdgcn_wmma_f32_16x16x32_f16(false, a, false, b,
                                               (short)0, c, false, false);
  }
  const int mb = tile_m * 16 + half * 8;
  float* out = s1 + (size_t)mb * HIDDEN + n;
#pragma unroll
  for (int r = 0; r < 8; ++r) out[(size_t)r * HIDDEN] = c[r];
}

// ---------------------------------------------------------------------------
// GEMM2: s2[N,48] = relu(agg1[N,128]) @ pad(W2)   (ReLU fused into A-load)
// Block = 96 threads = 3 waves = 3 column tiles. Grid = 6250.
// ---------------------------------------------------------------------------
__global__ __launch_bounds__(96) void gcn_gemm2_wmma(
    const float* __restrict__ agg1, const _Float16* __restrict__ w2p,
    float* __restrict__ s2) {
  const int lane   = threadIdx.x & 31;
  const int wave   = threadIdx.x >> 5;          // 0..2
  const int tile_m = blockIdx.x;
  const int mn     = lane & 15;
  const int half   = lane >> 4;
  const int khalf  = half * 8;
  const int m      = tile_m * 16 + mn;
  const int n      = wave * 16 + mn;
  const v16h* bp   = (const v16h*)w2p;

  v8f c = {};
#pragma unroll
  for (int kt = 0; kt < HIDDEN / 32; ++kt) {
    v16h a;
    const float* hr = agg1 + (size_t)m * HIDDEN + kt * 32 + khalf;
#pragma unroll
    for (int i = 0; i < 8; ++i) {
      float lo = hr[i], hi = hr[i + 16];
      a[i]     = (_Float16)(lo > 0.f ? lo : 0.f);   // fused ReLU
      a[i + 8] = (_Float16)(hi > 0.f ? hi : 0.f);
    }
    v16h b = bp[((size_t)kt * NCLASS_PAD + n) * 2 + half];
    c = __builtin_amdgcn_wmma_f32_16x16x32_f16(false, a, false, b,
                                               (short)0, c, false, false);
  }
  const int mb = tile_m * 16 + half * 8;
  float* out = s2 + (size_t)mb * NCLASS_PAD + n;
#pragma unroll
  for (int r = 0; r < 8; ++r) out[(size_t)r * NCLASS_PAD] = c[r];
}

// ---------------------------------------------------------------------------
// Accumulator init: agg[i] = bias[i % dim]  (fuses "+ b" into the scatter sum)
// ---------------------------------------------------------------------------
__global__ void gcn_init_bias(float* __restrict__ agg,
                              const float* __restrict__ bias,
                              int dim, int total) {
  int i = blockIdx.x * blockDim.x + threadIdx.x;
  if (i < total) agg[i] = bias[i % dim];
}

// ---------------------------------------------------------------------------
// SpMM layer 1: agg1[dst] += s1[src] * val, 128 features.
// One wave per edge (lane = float4 feature chunk, 32 chunks = 128 floats).
// Edge metadata forced into SGPRs via readfirstlane (wave-uniform by
// construction), keeping address math on the SALU. fp32 atomics resolve in
// L2 (agg1 is 51 MB, L2 is 192 MB).
// ---------------------------------------------------------------------------
__global__ void gcn_spmm1(const int* __restrict__ esrc,
                          const int* __restrict__ edst,
                          const float* __restrict__ eval,
                          const float* __restrict__ s1,
                          float* __restrict__ agg1) {
  int tid = blockIdx.x * blockDim.x + threadIdx.x;     // < E*32 (fits int32)
  int e = __builtin_amdgcn_readfirstlane(tid >> 5);    // wave-uniform edge id
  int q = threadIdx.x & 31;
  if (e >= N_EDGES) return;
  int   s = __builtin_amdgcn_readfirstlane(esrc[e]);
  int   d = __builtin_amdgcn_readfirstlane(edst[e]);
  float w = __int_as_float(__builtin_amdgcn_readfirstlane(__float_as_int(eval[e])));
  const float4 v = ((const float4*)(s1 + (size_t)s * HIDDEN))[q];
  float* dp = agg1 + (size_t)d * HIDDEN + q * 4;
  unsafeAtomicAdd(dp + 0, v.x * w);
  unsafeAtomicAdd(dp + 1, v.y * w);
  unsafeAtomicAdd(dp + 2, v.z * w);
  unsafeAtomicAdd(dp + 3, v.w * w);
}

// ---------------------------------------------------------------------------
// SpMM layer 2: agg2[dst] += s2[src] * val, 40 features (s2 stride 48).
// ---------------------------------------------------------------------------
__global__ void gcn_spmm2(const int* __restrict__ esrc,
                          const int* __restrict__ edst,
                          const float* __restrict__ eval,
                          const float* __restrict__ s2,
                          float* __restrict__ agg2) {
  int idx = blockIdx.x * blockDim.x + threadIdx.x;     // < E*40 = 64e6
  int e = idx / NCLASS;
  int f = idx - e * NCLASS;
  if (e >= N_EDGES) return;
  float m = s2[(size_t)esrc[e] * NCLASS_PAD + f] * eval[e];
  unsafeAtomicAdd(agg2 + (size_t)edst[e] * NCLASS + f, m);
}

// ---------------------------------------------------------------------------
// Row-wise log_softmax over 40 classes. One thread per node.
// ---------------------------------------------------------------------------
__global__ void gcn_logsoftmax(const float* __restrict__ agg2,
                               float* __restrict__ out) {
  int nidx = blockIdx.x * blockDim.x + threadIdx.x;
  if (nidx >= N_NODES) return;
  const float* row = agg2 + (size_t)nidx * NCLASS;
  float r[NCLASS];
  float mx = -INFINITY;
#pragma unroll
  for (int c = 0; c < NCLASS; ++c) { r[c] = row[c]; mx = fmaxf(mx, r[c]); }
  float s = 0.f;
#pragma unroll
  for (int c = 0; c < NCLASS; ++c) s += __expf(r[c] - mx);
  float lse = mx + __logf(s);
  float* o = out + (size_t)nidx * NCLASS;
#pragma unroll
  for (int c = 0; c < NCLASS; ++c) o[c] = r[c] - lse;
}

// ---------------------------------------------------------------------------
extern "C" void kernel_launch(void* const* d_in, const int* in_sizes, int n_in,
                              void* d_out, int out_size, void* d_ws, size_t ws_size,
                              hipStream_t stream) {
  const float* x    = (const float*)d_in[0];
  const int*   esrc = (const int*)  d_in[1];
  const int*   edst = (const int*)  d_in[2];
  const float* eval = (const float*)d_in[3];
  const float* W1   = (const float*)d_in[4];
  const float* b1   = (const float*)d_in[5];
  const float* W2   = (const float*)d_in[6];
  const float* b2   = (const float*)d_in[7];
  float* out = (float*)d_out;

  // Workspace layout (floats, all 32B-aligned offsets):
  //   s1 | agg1 | s2 | agg2 | W1pack(f16) | W2pack(f16)   (~138 MB total)
  float* s1   = (float*)d_ws;
  float* agg1 = s1   + (size_t)N_NODES * HIDDEN;
  float* s2   = agg1 + (size_t)N_NODES * HIDDEN;
  float* agg2 = s2   + (size_t)N_NODES * NCLASS_PAD;
  _Float16* w1p = (_Float16*)(agg2 + (size_t)N_NODES * NCLASS);
  _Float16* w2p = w1p + (NFEAT / 32) * HIDDEN * 32;      // 65536 halves

  // Weight pre-pack into WMMA B-fragment order (tiny, launch-invariant work)
  gcn_pack_w1<<<(NFEAT / 32) * HIDDEN * 32 / 256, 256, 0, stream>>>(W1, w1p);
  gcn_pack_w2<<<(HIDDEN / 32) * NCLASS_PAD * 32 / 256, 256, 0, stream>>>(W2, w2p);

  // Layer 1
  gcn_gemm1_wmma<<<N_NODES / 16, 256, 0, stream>>>(x, w1p, s1);
  gcn_init_bias<<<(N_NODES * HIDDEN) / 256, 256, 0, stream>>>(
      agg1, b1, HIDDEN, N_NODES * HIDDEN);
  gcn_spmm1<<<(N_EDGES * 32) / 256, 256, 0, stream>>>(esrc, edst, eval, s1, agg1);

  // Layer 2 (ReLU fused into GEMM2 A-load)
  gcn_gemm2_wmma<<<N_NODES / 16, 96, 0, stream>>>(agg1, w2p, s2);
  gcn_init_bias<<<(N_NODES * NCLASS + 255) / 256, 256, 0, stream>>>(
      agg2, b2, NCLASS, N_NODES * NCLASS);
  gcn_spmm2<<<(N_EDGES * NCLASS) / 256, 256, 0, stream>>>(esrc, edst, eval, s2, agg2);

  // Output
  gcn_logsoftmax<<<(N_NODES + 255) / 256, 256, 0, stream>>>(agg2, out);
}
